// SelfAttention_16088947491179
// MI455X (gfx1250) — compile-verified, matrix operand order checked by reading, not run on hardware
//
#include <hip/hip_runtime.h>

typedef __bf16 bf16;
typedef bf16  v16bf __attribute__((ext_vector_type(16)));
typedef bf16  v8bf  __attribute__((ext_vector_type(8)));
typedef float v8f   __attribute__((ext_vector_type(8)));
typedef float v4f   __attribute__((ext_vector_type(4)));

#define BT     8192
#define C_DIM  1024
#define D_HEAD 64
#define N_OUT  192   // q|k|v stacked output units

static __device__ __forceinline__ v16bf join16(v8bf lo, v8bf hi) {
  union { v16bf v; v8bf h[2]; } u;
  u.h[0] = lo; u.h[1] = hi;
  return u.v;
}

// Async memory->LDS 16-byte copy (CDNA5 GLOBAL_LOAD_ASYNC_TO_LDS_B128,
// tracked by ASYNCcnt). lds_ptr: flat pointer to __shared__ (low 32 bits are
// the LDS byte offset per ISA 10.2); gptr: global address.
static __device__ __forceinline__ void async_cp16(void* lds_ptr, const void* gptr) {
  unsigned lds_off = (unsigned)(uintptr_t)lds_ptr;
  asm volatile("global_load_async_to_lds_b128 %0, %1, off"
               :: "v"(lds_off), "v"(gptr)
               : "memory");
}

static __device__ __forceinline__ void wait_async0() {
  asm volatile("s_wait_asynccnt 0" ::: "memory");
}

// ---------------------------------------------------------------------------
// Kernel 1: fused QKV projection. 64 rows per WG, 4 waves x 16-row tiles.
// B-fragment loads are software-pipelined one tile ahead of the WMMA chain.
// Output: Qg (bf16, scaled by log2e/32), Kg (bf16), Vtg (bf16, transposed).
// ---------------------------------------------------------------------------
__global__ __launch_bounds__(128) void qkv_proj_kernel(
    const float* __restrict__ x,
    const float* __restrict__ Wq, const float* __restrict__ bq,
    const float* __restrict__ Wk, const float* __restrict__ bk,
    const float* __restrict__ Wv, const float* __restrict__ bv,
    bf16* __restrict__ Qg, bf16* __restrict__ Kg, bf16* __restrict__ Vtg)
{
  __shared__ __align__(16) bf16 Wc[N_OUT * 32];   // [out-unit][k-chunk] bf16

  const int tid  = threadIdx.x;
  const int wave = tid >> 5;
  const int lane = tid & 31;
  const int l16  = lane & 15;
  const int hi   = lane >> 4;
  const int kb   = hi ? 8 : 0;

  const int rowbase = blockIdx.x * 64 + wave * 16;
  const int arow    = rowbase + l16;      // A-fragment row for this lane

  v8f acc[12] = {};

  auto loadB = [&](int nt) -> v16bf {
    const bf16* wp = &Wc[(nt * 16 + l16) * 32];
    return join16(*(const v8bf*)(wp + kb), *(const v8bf*)(wp + 16 + kb));
  };

  for (int k0 = 0; k0 < C_DIM; k0 += 32) {
    __syncthreads();
    // cooperative f32->bf16 load of the 192x32 weight chunk
    for (int idx = tid; idx < N_OUT * 32; idx += 128) {
      int orow = idx >> 5, kc = idx & 31;
      float w;
      if (orow < 64)       w = Wq[orow * C_DIM + k0 + kc];
      else if (orow < 128) w = Wk[(orow - 64) * C_DIM + k0 + kc];
      else                 w = Wv[(orow - 128) * C_DIM + k0 + kc];
      Wc[idx] = (bf16)w;
    }
    __syncthreads();

    // A fragment: 16 f32 from this lane's x row, converted to bf16
    const float* xp = x + (size_t)arow * C_DIM + k0;
    v4f a0 = *(const v4f*)(xp + kb);
    v4f a1 = *(const v4f*)(xp + kb + 4);
    v4f a2 = *(const v4f*)(xp + 16 + kb);
    v4f a3 = *(const v4f*)(xp + 16 + kb + 4);
    v16bf afrag;
#pragma unroll
    for (int i = 0; i < 4; ++i) {
      afrag[i]      = (bf16)a0[i];
      afrag[4 + i]  = (bf16)a1[i];
      afrag[8 + i]  = (bf16)a2[i];
      afrag[12 + i] = (bf16)a3[i];
    }

    // software-pipelined B fragments: tile nt+1 loads behind wmma nt
    v16bf bfrag = loadB(0);
#pragma unroll
    for (int nt = 0; nt < 12; ++nt) {
      v16bf bnext = (nt < 11) ? loadB(nt + 1) : bfrag;
      acc[nt] = __builtin_amdgcn_wmma_f32_16x16x32_bf16(
          false, afrag, false, bfrag, (short)0, acc[nt], false, false);
      bfrag = bnext;
    }
  }

  // epilogue: bias, scale Q by log2(e)/sqrt(1024), scatter to Q/K/Vt
  const float qscale = 1.4426950408889634f / 32.0f;
#pragma unroll
  for (int nt = 0; nt < 12; ++nt) {
    const int o = nt * 16 + l16;
    float bias = (o < 64) ? bq[o] : (o < 128 ? bk[o - 64] : bv[o - 128]);
#pragma unroll
    for (int r = 0; r < 8; ++r) {
      const int row = rowbase + r + 8 * hi;
      float val = acc[nt][r] + bias;
      if (o < 64)        Qg [(size_t)row * D_HEAD + o]        = (bf16)(val * qscale);
      else if (o < 128)  Kg [(size_t)row * D_HEAD + (o - 64)] = (bf16)val;
      else               Vtg[(size_t)(o - 128) * BT + row]    = (bf16)val;
    }
  }
}

// ---------------------------------------------------------------------------
// Kernel 2: flash attention over the full 8192 token sequence.
// Per WG: 64 query rows (4 waves x 16). Per 32-key block: 4 WMMA for S,
// online softmax, 4 WMMA for P.V. K/Vt blocks are staged into LDS with
// double-buffered GLOBAL_LOAD_ASYNC_TO_LDS_B128 so the next block's loads
// overlap the current block's WMMAs + softmax. All fragment ds_loads for a
// WMMA burst are issued before the burst.
// ---------------------------------------------------------------------------
__global__ __launch_bounds__(128) void flash_attn_kernel(
    const bf16* __restrict__ Qg, const bf16* __restrict__ Kg,
    const bf16* __restrict__ Vtg, float* __restrict__ Yg)
{
  __shared__ __align__(16) bf16 Kblk[2][32 * 64];    // [buf][key][d]
  __shared__ __align__(16) bf16 Vblk[2][64 * 32];    // [buf][d][key] (transposed)
  __shared__ __align__(16) bf16 Pblk[4][16 * 32];    // per-wave P scratch

  const int tid  = threadIdx.x;
  const int wave = tid >> 5;
  const int lane = tid & 31;
  const int l16  = lane & 15;
  const int hi   = lane >> 4;
  const int kb   = hi ? 8 : 0;
  const int rowbase = blockIdx.x * 64 + wave * 16;

  // Q fragments: K-dim 64 = two chunks of 32
  v16bf qf[2];
  {
    const bf16* qp = Qg + (size_t)(rowbase + l16) * D_HEAD;
#pragma unroll
    for (int kk = 0; kk < 2; ++kk)
      qf[kk] = join16(*(const v8bf*)(qp + kk * 32 + kb),
                      *(const v8bf*)(qp + kk * 32 + 16 + kb));
  }

  v8f O[4] = {};
  float m[8], l[8];
#pragma unroll
  for (int r = 0; r < 8; ++r) { m[r] = -1e30f; l[r] = 0.0f; }

  // issue async stage of a 32-key block into LDS buffer `buf`
  auto stage = [&](int kb0, int buf) {
    // K block: contiguous 4KB (32 rows x 64 bf16)
    const char* ks = (const char*)(Kg + (size_t)kb0 * D_HEAD);
    char* kd = (char*)Kblk[buf];
    async_cp16(kd + tid * 16,         ks + tid * 16);
    async_cp16(kd + (tid + 128) * 16, ks + (tid + 128) * 16);
    // Vt block: 64 rows x 32 keys (64B per row)
    char* vd = (char*)Vblk[buf];
#pragma unroll
    for (int e0 = 0; e0 < 256; e0 += 128) {
      int e = e0 + tid;
      int d = e >> 2, q4 = e & 3;
      async_cp16(vd + e * 16, (const char*)(Vtg + (size_t)d * BT + kb0 + q4 * 8));
    }
  };

  stage(0, 0);

  for (int blk = 0; blk < BT / 32; ++blk) {
    const int cur = blk & 1;
    wait_async0();        // current buffer's async loads (this wave) complete
    __syncthreads();      // all waves' loads done; all done reading other buf
    if (blk + 1 < BT / 32) stage((blk + 1) * 32, cur ^ 1);

    // preload all four K fragments, then run the S WMMA burst
    v16bf kf[2][2];
#pragma unroll
    for (int ct = 0; ct < 2; ++ct) {
      const bf16* kp = Kblk[cur] + (ct * 16 + l16) * D_HEAD;
#pragma unroll
      for (int kk = 0; kk < 2; ++kk)
        kf[ct][kk] = join16(*(const v8bf*)(kp + kk * 32 + kb),
                            *(const v8bf*)(kp + kk * 32 + 16 + kb));
    }
    v8f S[2];
#pragma unroll
    for (int ct = 0; ct < 2; ++ct) {
      v8f c = {};
      c = __builtin_amdgcn_wmma_f32_16x16x32_bf16(
          false, qf[0], false, kf[ct][0], (short)0, c, false, false);
      c = __builtin_amdgcn_wmma_f32_16x16x32_bf16(
          false, qf[1], false, kf[ct][1], (short)0, c, false, false);
      S[ct] = c;
    }

    // online softmax (log2-domain: scale already folded into Q)
#pragma unroll
    for (int r = 0; r < 8; ++r) {
      float mx = fmaxf(S[0][r], S[1][r]);
      mx = fmaxf(mx, __shfl_xor(mx, 1, 32));
      mx = fmaxf(mx, __shfl_xor(mx, 2, 32));
      mx = fmaxf(mx, __shfl_xor(mx, 4, 32));
      mx = fmaxf(mx, __shfl_xor(mx, 8, 32));
      const float mn    = fmaxf(m[r], mx);
      const float alpha = __builtin_amdgcn_exp2f(m[r] - mn);
      m[r] = mn;
      const float p0 = __builtin_amdgcn_exp2f(S[0][r] - mn);
      const float p1 = __builtin_amdgcn_exp2f(S[1][r] - mn);
      const int prow = r + 8 * hi;
      Pblk[wave][prow * 32 + l16]      = (bf16)p0;   // ct=0 keys
      Pblk[wave][prow * 32 + 16 + l16] = (bf16)p1;   // ct=1 keys
      float s = p0 + p1;
      s += __shfl_xor(s, 1, 32);
      s += __shfl_xor(s, 2, 32);
      s += __shfl_xor(s, 4, 32);
      s += __shfl_xor(s, 8, 32);
      l[r] = l[r] * alpha + s;
#pragma unroll
      for (int nt = 0; nt < 4; ++nt) O[nt][r] *= alpha;
    }

    // in-wave LDS RAW: make P visible before fragment reload
    asm volatile("s_wait_dscnt 0" ::: "memory");

    // preload P fragment + all four V fragments, then the P.V WMMA burst
    const bf16* pp = &Pblk[wave][l16 * 32];
    v16bf pf = join16(*(const v8bf*)(pp + kb), *(const v8bf*)(pp + 16 + kb));
    v16bf vf[4];
#pragma unroll
    for (int nt = 0; nt < 4; ++nt) {
      const bf16* vp = Vblk[cur] + (nt * 16 + l16) * 32;
      vf[nt] = join16(*(const v8bf*)(vp + kb), *(const v8bf*)(vp + 16 + kb));
    }
#pragma unroll
    for (int nt = 0; nt < 4; ++nt)
      O[nt] = __builtin_amdgcn_wmma_f32_16x16x32_bf16(
          false, pf, false, vf[nt], (short)0, O[nt], false, false);
  }

  // normalize and write y (BT x 64, f32)
#pragma unroll
  for (int r = 0; r < 8; ++r) {
    const float inv = 1.0f / l[r];
    const int row = rowbase + r + 8 * hi;
#pragma unroll
    for (int nt = 0; nt < 4; ++nt)
      Yg[(size_t)row * D_HEAD + nt * 16 + l16] = O[nt][r] * inv;
  }
}

// ---------------------------------------------------------------------------
// Kernel 3: tile the 64-dim head result 16x across the embedding dim.
// ---------------------------------------------------------------------------
__global__ __launch_bounds__(256) void head_bcast_kernel(
    const float* __restrict__ Yg, float* __restrict__ out)
{
  const size_t i = (size_t)blockIdx.x * 256 + threadIdx.x;  // float4 index
  const size_t row = i >> 8;          // 256 float4 per output row
  const int    c4  = (int)(i & 255);
  ((v4f*)out)[i] = ((const v4f*)Yg)[row * 16 + (c4 & 15)];
}

// ---------------------------------------------------------------------------
extern "C" void kernel_launch(void* const* d_in, const int* in_sizes, int n_in,
                              void* d_out, int out_size, void* d_ws, size_t ws_size,
                              hipStream_t stream) {
  const float* x  = (const float*)d_in[0];
  const float* Wq = (const float*)d_in[1];
  const float* bq = (const float*)d_in[2];
  const float* Wk = (const float*)d_in[3];
  const float* bk = (const float*)d_in[4];
  const float* Wv = (const float*)d_in[5];
  const float* bv = (const float*)d_in[6];

  char* ws = (char*)d_ws;
  bf16*  Qg  = (bf16*)(ws);                     // 1 MB
  bf16*  Kg  = (bf16*)(ws + (1u << 20));        // 1 MB
  bf16*  Vtg = (bf16*)(ws + (2u << 20));        // 1 MB
  float* Yg  = (float*)(ws + (3u << 20));       // 2 MB
  float* out = (float*)d_out;

  qkv_proj_kernel<<<128, 128, 0, stream>>>(x, Wq, bq, Wk, bk, Wv, bv, Qg, Kg, Vtg);
  flash_attn_kernel<<<128, 128, 0, stream>>>(Qg, Kg, Vtg, Yg);
  head_bcast_kernel<<<8192, 256, 0, stream>>>(Yg, out);
}
